// LorentzLinear_1400159338518
// MI455X (gfx1250) — matrix operand order repaired and checked by reading, not compiled, not run on hardware
//
#include <hip/hip_runtime.h>

// LorentzLinear fused GEMM + epilogue for gfx1250 (MI455X).
// Y = X(65536x512) * W^T(512x512) + b, then row-wise Lorentz normalization.
//   - bf16 WMMA (v_wmma_f32_16x16x32_bf16), fp32 accumulate, fused epilogue
//   - W pre-packed to bf16 in d_ws, streamed to LDS via global_load_async_to_lds_b128
//     (ASYNCcnt / s_wait_asynccnt), double-buffered against WMMA compute.

#define N_TOTAL 65536
#define KD      512
#define ND      512
#define BLK_M   64
#define KC      32
#define KITERS  (KD / KC)

typedef __attribute__((ext_vector_type(16))) __bf16 v16bf;
typedef __attribute__((ext_vector_type(8)))  float  v8f;

union BFrag {
  unsigned u[8];
  uint4    q[2];
  v16bf    v;
};

__device__ __forceinline__ unsigned pack2_bf16(float a, float b) {
  union { float f; unsigned u; } ua, ub;
  ua.f = a; ub.f = b;
  unsigned ra = ua.u + 0x7FFFu + ((ua.u >> 16) & 1u);  // round-to-nearest-even
  unsigned rb = ub.u + 0x7FFFu + ((ub.u >> 16) & 1u);
  return (ra >> 16) | (rb & 0xFFFF0000u);
}

// ---- prep: pack W (512x512 f32, k-major) into bf16 pairs in workspace ----
__global__ void wconvert_kernel(const float* __restrict__ Wm,
                                unsigned* __restrict__ Wbf) {
  int i = blockIdx.x * blockDim.x + threadIdx.x;   // u32 id, 0..131071
  float2 v = reinterpret_cast<const float2*>(Wm)[i];
  Wbf[i] = pack2_bf16(v.x, v.y);
}

template <bool ASYNC_W>
__global__ __launch_bounds__(256)
void lorentz_linear_kernel(const float* __restrict__ x,
                           const float* __restrict__ Wm,
                           const unsigned* __restrict__ Wbf,
                           const float* __restrict__ bias,
                           const float* __restrict__ scale_p,
                           float* __restrict__ out)
{
  // Double-buffered bf16 tiles (packed u32 pairs, K-major).
  __shared__ __align__(16) unsigned As_u[2][BLK_M][KC / 2];  // 2 x 4 KB
  __shared__ __align__(16) unsigned Ws_u[2][ND][KC / 2];     // 2 x 32 KB
  __shared__ float rowsum[BLK_M];
  __shared__ float y0s[BLK_M];
  __shared__ float timef[BLK_M];
  __shared__ float sfac[BLK_M];

  const int  tid  = threadIdx.x;
  const int  lane = tid & 31;       // wave32
  const int  wave = tid >> 5;       // 8 waves
  const int  wm   = wave >> 1;      // 0..3 : 16-row M tile
  const int  wn   = wave & 1;       // 0..1 : 256-col N half
  const int  lrow = lane & 15;
  const bool hi   = lane >= 16;
  const int  m0   = blockIdx.x * BLK_M;

  if (tid < BLK_M) rowsum[tid] = 0.0f;

  v8f acc[16];
  #pragma unroll
  for (int t = 0; t < 16; ++t)
    #pragma unroll
    for (int r = 0; r < 8; ++r) acc[t][r] = 0.0f;

  // ---- staging: X tile (64x32 f32 -> bf16, VALU pack) ----
  auto stageX = [&](int buf, int k) {
    #pragma unroll
    for (int i = 0; i < 2; ++i) {
      int f   = tid + i * 256;            // float4 id 0..511
      int row = f >> 3;
      int c4  = (f & 7) << 2;
      float4 v = *reinterpret_cast<const float4*>(
          &x[(size_t)(m0 + row) * KD + k + c4]);
      As_u[buf][row][(c4 >> 1)]     = pack2_bf16(v.x, v.y);
      As_u[buf][row][(c4 >> 1) + 1] = pack2_bf16(v.z, v.w);
    }
  };
  // ---- staging: W tile (512x32 bf16 = 32 KB) ----
  auto stageW = [&](int buf, int k) {
    if (ASYNC_W) {
      // Raw async copy from pre-packed bf16 W: 8 x B128 per thread,
      // tracked on ASYNCcnt, overlaps with WMMA on the other buffer.
      unsigned ldsbase = (unsigned)(unsigned long long)(void*)&Ws_u[buf][0][0];
      const char* gbase = reinterpret_cast<const char*>(Wbf) + (size_t)k * 2;
      #pragma unroll
      for (int i = 0; i < 8; ++i) {
        int c = tid + i * 256;            // B128 chunk id 0..2047
        unsigned o    = (unsigned)(c >> 2);
        unsigned part = (unsigned)(c & 3);
        unsigned voff = o * 1024u + part * 16u;   // W row stride 512*2 B
        unsigned la   = ldsbase + o * 64u + part * 16u;
        asm volatile("global_load_async_to_lds_b128 %0, %1, %2"
                     :: "v"(la), "v"(voff), "s"(gbase) : "memory");
      }
    } else {
      // Fallback: convert fp32 W on the fly.
      #pragma unroll
      for (int i = 0; i < 16; ++i) {
        int f  = tid + i * 256;           // float4 id 0..4095
        int o  = f >> 3;
        int c4 = (f & 7) << 2;
        float4 v = *reinterpret_cast<const float4*>(
            &Wm[(size_t)o * KD + k + c4]);
        Ws_u[buf][o][(c4 >> 1)]     = pack2_bf16(v.x, v.y);
        Ws_u[buf][o][(c4 >> 1) + 1] = pack2_bf16(v.z, v.w);
      }
    }
  };

  // prologue: fill buffer 0
  stageX(0, 0);
  stageW(0, 0);
  if (ASYNC_W) asm volatile("s_wait_asynccnt 0" ::: "memory");
  __syncthreads();

  int cur = 0;
  for (int it = 0; it < KITERS; ++it) {
    // kick next tile into the other buffer (overlaps with WMMA below)
    if (it + 1 < KITERS) {
      stageX(cur ^ 1, (it + 1) * KC);
      stageW(cur ^ 1, (it + 1) * KC);
    }

    // A fragment (16x32 bf16, CDNA5 layout):
    // lane<16 : row=lane,    K = {0..7, 16..23}
    // lane>=16: row=lane-16, K = {8..15, 24..31}
    BFrag a;
    {
      int arow = wm * 16 + lrow;
      int kb2  = hi ? 4 : 0;
      a.q[0] = *reinterpret_cast<const uint4*>(&As_u[cur][arow][kb2]);
      a.q[1] = *reinterpret_cast<const uint4*>(&As_u[cur][arow][kb2 + 8]);
    }

    // B fragments (32x16 bf16): lane<16 col=lane K{0..7,16..23}, lane>=16 K{8..15,24..31}.
    // Software-pipelined: load b[t+1] before wmma[t].
    const int kb2b = hi ? 4 : 0;
    BFrag bcur;
    {
      int o = wn * 256 + lrow;
      bcur.q[0] = *reinterpret_cast<const uint4*>(&Ws_u[cur][o][kb2b]);
      bcur.q[1] = *reinterpret_cast<const uint4*>(&Ws_u[cur][o][kb2b + 8]);
    }
    #pragma unroll
    for (int t = 0; t < 16; ++t) {
      BFrag bnext;
      if (t < 15) {
        int o = wn * 256 + (t + 1) * 16 + lrow;
        bnext.q[0] = *reinterpret_cast<const uint4*>(&Ws_u[cur][o][kb2b]);
        bnext.q[1] = *reinterpret_cast<const uint4*>(&Ws_u[cur][o][kb2b + 8]);
      }
      acc[t] = __builtin_amdgcn_wmma_f32_16x16x32_bf16(
          false, a.v, false, bcur.v, (short)0, acc[t], false, false);
      if (t < 15) bcur = bnext;
    }

    if (ASYNC_W && it + 1 < KITERS)
      asm volatile("s_wait_asynccnt 0" ::: "memory");
    __syncthreads();
    cur ^= 1;
  }

  // ---- epilogue: bias add, per-row sum of squares over cols 1..511 ----
  float ss[8];
  float y0v[8];
  #pragma unroll
  for (int r = 0; r < 8; ++r) { ss[r] = 0.0f; y0v[r] = 0.0f; }

  #pragma unroll
  for (int t = 0; t < 16; ++t) {
    int   o  = wn * 256 + t * 16 + lrow;
    float bv = bias[o];
    bool  c0 = (o == 0);
    #pragma unroll
    for (int r = 0; r < 8; ++r) {
      float c = acc[t][r] + bv;
      acc[t][r] = c;
      float c2 = c * c;
      if (c0) { y0v[r] = c; c2 = 0.0f; }  // column 0 excluded from norm
      ss[r] += c2;
    }
  }
  // butterfly reduce across the 16 lanes sharing each row
  #pragma unroll
  for (int r = 0; r < 8; ++r) {
    #pragma unroll
    for (int off = 8; off > 0; off >>= 1)
      ss[r] += __shfl_xor(ss[r], off, 16);
  }
  if (lrow == 0) {
    #pragma unroll
    for (int r = 0; r < 8; ++r) {
      int row = wm * 16 + r + (hi ? 8 : 0);
      atomicAdd(&rowsum[row], ss[r]);       // ds_add_f32
      if (wn == 0) y0s[row] = y0v[r];
    }
  }
  __syncthreads();

  if (tid < BLK_M) {
    float es  = __expf(scale_p[0]);
    float y0  = y0s[tid];
    float tm  = es / (1.0f + __expf(-y0)) + 1.1f;    // sigmoid(y0)*e^s + 1.1
    float den = fmaxf(rowsum[tid], 1e-8f);
    timef[tid] = tm;
    sfac[tid]  = __fsqrt_rn((tm * tm - 1.0f) / den); // tm >= 1.1 -> non-negative
  }
  __syncthreads();

  // ---- scaled stores (C layout: VGPR r -> row r / r+8, lane%16 -> col) ----
  #pragma unroll
  for (int t = 0; t < 16; ++t) {
    int o = wn * 256 + t * 16 + lrow;
    #pragma unroll
    for (int r = 0; r < 8; ++r) {
      int   row = wm * 16 + r + (hi ? 8 : 0);
      float v   = acc[t][r] * sfac[row];
      if (o == 0) v = timef[row];
      out[(size_t)(m0 + row) * ND + o] = v;
    }
  }
}

extern "C" void kernel_launch(void* const* d_in, const int* in_sizes, int n_in,
                              void* d_out, int out_size, void* d_ws, size_t ws_size,
                              hipStream_t stream) {
  const float* x     = (const float*)d_in[0];
  const float* Wm    = (const float*)d_in[1];
  const float* bias  = (const float*)d_in[2];
  const float* scale = (const float*)d_in[3];
  float*       out   = (float*)d_out;

  dim3 grid(N_TOTAL / BLK_M);   // 1024 blocks
  dim3 block(256);              // 8 wave32 waves

  const size_t wbf_bytes = (size_t)(ND * KD / 2) * sizeof(unsigned);  // 512 KB
  if (d_ws != nullptr && ws_size >= wbf_bytes) {
    unsigned* Wbf = (unsigned*)d_ws;
    hipLaunchKernelGGL(wconvert_kernel, dim3(ND * KD / 2 / 256), dim3(256),
                       0, stream, Wm, Wbf);
    hipLaunchKernelGGL((lorentz_linear_kernel<true>), grid, block, 0, stream,
                       x, Wm, Wbf, bias, scale, out);
  } else {
    hipLaunchKernelGGL((lorentz_linear_kernel<false>), grid, block, 0, stream,
                       x, Wm, nullptr, bias, scale, out);
  }
}
